// LinearPatchMerger_37469294690389
// MI455X (gfx1250) — matrix operand order repaired and checked by reading, not compiled, not run on hardware
//
#include <hip/hip_runtime.h>
#include <hip/hip_bf16.h>

typedef __attribute__((ext_vector_type(16))) __bf16 bf16x16;
typedef __attribute__((ext_vector_type(8)))  __bf16 bf16x8;
typedef __attribute__((ext_vector_type(8)))  float  f32x8;
typedef __attribute__((ext_vector_type(4)))  float  f32x4;

#define D_IN    1024   // channels per patch
#define KP      4096   // D * MERGE^2 (GEMM K)
#define N_OUT   1024   // GEMM N
#define BM      128
#define BN      64
#define BK      32
#define KITER   (KP / BK)
#define LDK     40     // padded LDS row stride in bf16 (80 B, 16 B aligned)
#define THREADS 256

// ---------------------------------------------------------------------------
// Prep: repack W (1024 x 4096 fp32, k = d*4+sub) into split-bf16 Wp with the
// K axis permuted to k' = sub*1024 + d.  Under this consistent permutation the
// gathered A tiles AND the B tiles are fully contiguous in the main GEMM.
// ---------------------------------------------------------------------------
__global__ __launch_bounds__(256)
void pack_w_kernel(const float* __restrict__ W,
                   __bf16* __restrict__ Whi,
                   __bf16* __restrict__ Wlo) {
  int idx = blockIdx.x * blockDim.x + threadIdx.x;   // n*4096 + k'
  int n   = idx >> 12;
  int kp  = idx & 4095;
  int sub = kp >> 10;                                // which of 4 merged patches
  int d   = kp & 1023;                               // channel
  float w = W[(n << 12) + (d << 2) + sub];
  __bf16 hi = (__bf16)w;
  __bf16 lo = (__bf16)(w - (float)hi);
  Whi[idx] = hi;
  Wlo[idx] = lo;
}

// A fragment (16x32 bf16, M x K): lane L<16 -> row L, K {0..7, 16..23};
// lane L>=16 -> row L-16, K {8..15, 24..31}.
__device__ __forceinline__ bf16x16 frag_a(const __bf16* base, int r0, int lane) {
  int r  = r0 + (lane & 15);
  int kb = (lane >> 4) << 3;                 // 0 or 8
  const __bf16* p = base + r * LDK + kb;
  bf16x8 v0 = *(const bf16x8*)(p);           // K = kb .. kb+7
  bf16x8 v1 = *(const bf16x8*)(p + 16);      // K = kb+16 .. kb+23
  return __builtin_shufflevector(v0, v1, 0,1,2,3,4,5,6,7,8,9,10,11,12,13,14,15);
}

// B fragment (32x16 bf16, K x N): lane L<16 -> col L, K {0..7, 16..23};
// lane L>=16 -> col L-16, K {8..15, 24..31}.
__device__ __forceinline__ bf16x16 frag_b(const __bf16* base, int r0, int lane) {
  int n  = r0 + (lane & 15);
  int kb = (lane >> 4) << 3;                 // 0 or 8
  const __bf16* p = base + n * LDK + kb;
  bf16x8 v0 = *(const bf16x8*)(p);
  bf16x8 v1 = *(const bf16x8*)(p + 16);
  return __builtin_shufflevector(v0, v1, 0,1,2,3,4,5,6,7,8,9,10,11,12,13,14,15);
}

// ---------------------------------------------------------------------------
// Fused gather (patch merge) + split-bf16 GEMM, double-buffered LDS pipeline:
// stage tile k+1 while computing tile k; single barrier per K-iteration.
// ---------------------------------------------------------------------------
__global__ __launch_bounds__(THREADS)
void merger_gemm_kernel(const float* __restrict__ X,
                        const __bf16* __restrict__ Whi,
                        const __bf16* __restrict__ Wlo,
                        float* __restrict__ out) {
  __shared__ __bf16 sAhi[2][BM * LDK];
  __shared__ __bf16 sAlo[2][BM * LDK];
  __shared__ __bf16 sBhi[2][BN * LDK];
  __shared__ __bf16 sBlo[2][BN * LDK];

  const int tid  = threadIdx.x;
  const int lane = tid & 31;
  const int wave = tid >> 5;
  const int wm   = wave & 3;                 // 4 waves along M (32 rows each)
  const int wn   = wave >> 2;                // 2 waves along N (32 cols each)
  const int m0   = blockIdx.y * BM;
  const int n0   = blockIdx.x * BN;

  // A staging geometry: 2 threads per merged row, 16 floats each
  const int arow  = tid >> 1;
  const int ahalf = tid & 1;
  const int m     = m0 + arow;
  const int img   = m >> 12;                 // 4096 merged rows per image
  const int I     = (m >> 6) & 63;
  const int J     = m & 63;

  // B staging geometry: part(hi/lo) x 64 rows x 2 halves = 256 threads
  const int bpart = tid >> 7;
  const int brow  = (tid & 127) >> 1;
  const int bhalf = tid & 1;
  const __bf16* wsrc = (bpart ? Wlo : Whi) + (size_t)(n0 + brow) * KP + bhalf * 16;

  // Stage K-tile `kb` into LDS buffer `p`
  auto stage = [&](int kb, int p) {
    const int sub = kb >> 5;                 // k' = sub*1024 + d; BK | 1024
    const int d0  = (kb << 5) & 1023;
    // ---- A: gather merge-permuted source row, fp32 -> bf16 hi/lo split ----
    const int srow = (img << 14) + (((I << 1) + (sub >> 1)) << 7)
                   + ((J << 1) + (sub & 1));
    const float* psrc = X + (size_t)srow * D_IN + d0 + ahalf * 16;
    float f[16];
    *(f32x4*)(f + 0)  = *(const f32x4*)(psrc + 0);
    *(f32x4*)(f + 4)  = *(const f32x4*)(psrc + 4);
    *(f32x4*)(f + 8)  = *(const f32x4*)(psrc + 8);
    *(f32x4*)(f + 12) = *(const f32x4*)(psrc + 12);
    __bf16 h[16], l[16];
    #pragma unroll
    for (int i = 0; i < 16; ++i) {
      __bf16 hh = (__bf16)f[i];
      h[i] = hh;
      l[i] = (__bf16)(f[i] - (float)hh);
    }
    __bf16* ah = &sAhi[p][arow * LDK + ahalf * 16];
    __bf16* al = &sAlo[p][arow * LDK + ahalf * 16];
    *(bf16x8*)(ah)     = *(bf16x8*)(h);
    *(bf16x8*)(ah + 8) = *(bf16x8*)(h + 8);
    *(bf16x8*)(al)     = *(bf16x8*)(l);
    *(bf16x8*)(al + 8) = *(bf16x8*)(l + 8);
    // ---- B: pre-split bf16 weights, contiguous copy (L2-resident) ----
    const __bf16* pw = wsrc + (kb << 5);
    __bf16* bd = (bpart ? &sBlo[p][0] : &sBhi[p][0]) + brow * LDK + bhalf * 16;
    *(bf16x8*)(bd)     = *(const bf16x8*)(pw);
    *(bf16x8*)(bd + 8) = *(const bf16x8*)(pw + 8);
  };

  f32x8 acc[2][2] = {};

  stage(0, 0);
  for (int kb = 0; kb < KITER; ++kb) {
    const int p = kb & 1;
    __syncthreads();                 // staged tile `kb` visible; prior reads done
    if (kb + 1 < KITER) stage(kb + 1, p ^ 1);

    bf16x16 Ah[2], Al[2], Bh[2], Bl[2];
    #pragma unroll
    for (int i = 0; i < 2; ++i) {
      Ah[i] = frag_a(&sAhi[p][0], wm * 32 + i * 16, lane);
      Al[i] = frag_a(&sAlo[p][0], wm * 32 + i * 16, lane);
    }
    #pragma unroll
    for (int j = 0; j < 2; ++j) {
      Bh[j] = frag_b(&sBhi[p][0], wn * 32 + j * 16, lane);
      Bl[j] = frag_b(&sBlo[p][0], wn * 32 + j * 16, lane);
    }
    #pragma unroll
    for (int i = 0; i < 2; ++i) {
      #pragma unroll
      for (int j = 0; j < 2; ++j) {
        acc[i][j] = __builtin_amdgcn_wmma_f32_16x16x32_bf16(
            false, Ah[i], false, Bh[j], (short)0, acc[i][j], false, false);
        acc[i][j] = __builtin_amdgcn_wmma_f32_16x16x32_bf16(
            false, Al[i], false, Bh[j], (short)0, acc[i][j], false, false);
        acc[i][j] = __builtin_amdgcn_wmma_f32_16x16x32_bf16(
            false, Ah[i], false, Bl[j], (short)0, acc[i][j], false, false);
      }
    }
  }

  // Epilogue: 16x16 f32 C layout -> VGPR v holds M = v (+8 for lanes 16..31)
  const int rl = (lane >> 4) << 3;
  const int nc = lane & 15;
  #pragma unroll
  for (int i = 0; i < 2; ++i) {
    #pragma unroll
    for (int j = 0; j < 2; ++j) {
      const int rbase = m0 + wm * 32 + i * 16 + rl;
      const int cbase = n0 + wn * 32 + j * 16 + nc;
      #pragma unroll
      for (int v = 0; v < 8; ++v)
        out[(size_t)(rbase + v) * N_OUT + cbase] = acc[i][j][v];
    }
  }
}

extern "C" void kernel_launch(void* const* d_in, const int* in_sizes, int n_in,
                              void* d_out, int out_size, void* d_ws, size_t ws_size,
                              hipStream_t stream) {
  const float* X = (const float*)d_in[0];      // image_features (1,131072,1024)
  // d_in[1] = image_sizes: all 1792 -> 128x128 patches, 64x64 merged per image
  const float* W = (const float*)d_in[2];      // (1024, 4096)
  float* out = (float*)d_out;                  // (1, 32768, 1024) fp32

  __bf16* Whi = (__bf16*)d_ws;                              // 8 MB
  __bf16* Wlo = Whi + (size_t)N_OUT * KP;                   // 8 MB
  pack_w_kernel<<<(N_OUT * KP) / 256, 256, 0, stream>>>(W, Whi, Wlo);

  const int Mtot = 8 * 64 * 64;                             // 32768 merged rows
  dim3 grid(N_OUT / BN, Mtot / BM);                         // x fastest -> A reuse in L2
  merger_gemm_kernel<<<grid, THREADS, 0, stream>>>(X, Whi, Wlo, out);

  (void)in_sizes; (void)n_in; (void)out_size; (void)ws_size;
}
